// PointTransformerLayer_79164837200444
// MI455X (gfx1250) — matrix operand order repaired and checked by reading, not compiled, not run on hardware
//
#include <hip/hip_runtime.h>
#include <hip/hip_bf16.h>

// ---------------------------------------------------------------------------
// PointTransformer layer for MI455X (gfx1250, wave32, WMMA).
//
// Plan:
//   Pass 1 (qkv_wmma): XQ/XK/XV = x @ {Wq,Wk,Wv}^T + b  -> 48 MB scratch.
//     Dense GEMM done with v_wmma_f32_16x16x32_bf16 (2 WMMAs per 16x16 tile).
//   Pass 2 (attn_kernel): per point, gather 16 neighbors (rows hit the 192 MB
//     L2 since XK/XV are only 16 MB each), positional MLP, relation, the
//     16x64 @ 64x8 reduction as one WMMA pair, 8x8 GEMM + softmax, weighted sum.
// ---------------------------------------------------------------------------

#define NPTS  65536
#define CC    64
#define NSAMP 16
#define CSS   8
#define EPSV  1e-5f

typedef __attribute__((ext_vector_type(16))) __bf16 v16bf;
typedef __attribute__((ext_vector_type(8)))  float  v8f;

// ---------------------------------------------------------------------------
// Kernel 1: fused QKV projection.  One wave (32 threads) per 16-row tile of x.
// A order used here (16x32, contiguous k): lanes 0-15 = row M=lane, K 0..15;
//                           lanes 16-31 = row M=lane-16, K 16..31.
// B order used here (32x16, contiguous k): lanes 0-15 = col N=lane, K 0..15;
//                           lanes 16-31 = col N=lane-16, K 16..31.
// D layout (f32 16x16): elem r -> lanes 0-15: (M=r, N=lane); lanes 16-31: (M=r+8).
// ---------------------------------------------------------------------------
__global__ __launch_bounds__(32) void qkv_wmma(
    const float* __restrict__ x,
    const float* __restrict__ Wq, const float* __restrict__ bq,
    const float* __restrict__ Wk, const float* __restrict__ bk,
    const float* __restrict__ Wv, const float* __restrict__ bv,
    float* __restrict__ XQ, float* __restrict__ XK, float* __restrict__ XV)
{
    const int lane = threadIdx.x;          // 0..31
    const int tile = blockIdx.x;           // NPTS/16 tiles
    const int row  = lane & 15;
    const int grow = tile * 16 + row;
    const int kb0  = (lane < 16) ? 0 : 16; // A K-base
    const float* xr = x + (size_t)grow * CC;

    // Build A (x tile) for K=0..31 and K=32..63
    v16bf a0, a1;
#pragma unroll
    for (int e = 0; e < 16; ++e) {
        const int k = kb0 + e;
        a0[e] = (__bf16)xr[k];
        a1[e] = (__bf16)xr[32 + k];
    }

    const int col = lane & 15;
    const int kbB = (lane < 16) ? 0 : 16;  // B K-base
    const int mb  = (lane < 16) ? 0 : 8;   // D row base

    const float* Ws[3] = {Wq, Wk, Wv};
    const float* Bs[3] = {bq, bk, bv};
    float*       Os[3] = {XQ, XK, XV};

#pragma unroll
    for (int m = 0; m < 3; ++m) {
#pragma unroll
        for (int n4 = 0; n4 < 4; ++n4) {
            const int colg = n4 * 16 + col;              // output channel
            const float* wr = Ws[m] + (size_t)colg * CC; // W row = out channel
            v16bf b0, b1;
#pragma unroll
            for (int e = 0; e < 16; ++e) {
                b0[e] = (__bf16)wr[kbB + e];
                b1[e] = (__bf16)wr[32 + kbB + e];
            }
            v8f acc = {};
            acc = __builtin_amdgcn_wmma_f32_16x16x32_bf16(
                      false, a0, false, b0, (short)0, acc, false, false);
            acc = __builtin_amdgcn_wmma_f32_16x16x32_bf16(
                      false, a1, false, b1, (short)0, acc, false, false);
            const float bb = Bs[m][colg];
            float* O = Os[m];
#pragma unroll
            for (int r = 0; r < 8; ++r)
                O[(size_t)(tile * 16 + mb + r) * CC + colg] = acc[r] + bb;
        }
    }
}

// ---------------------------------------------------------------------------
// Kernel 2: per-point attention.  64 threads (2 waves) per point.
// thread = channel c (0..63).  Wave 0 runs the 16x64 @ 64x8 WMMA.
// ---------------------------------------------------------------------------
__global__ __launch_bounds__(64) void attn_kernel(
    const float* __restrict__ p,   const int* __restrict__ idx,
    const float* __restrict__ Wp1, const float* __restrict__ bp1,
    const float* __restrict__ pg,  const float* __restrict__ pb,
    const float* __restrict__ pm,  const float* __restrict__ pv,
    const float* __restrict__ Wp2, const float* __restrict__ bp2,
    const float* __restrict__ wg1, const float* __restrict__ wb1,
    const float* __restrict__ wm1, const float* __restrict__ wv1,
    const float* __restrict__ Ww1, const float* __restrict__ bw1,
    const float* __restrict__ wg2, const float* __restrict__ wb2,
    const float* __restrict__ wm2, const float* __restrict__ wv2,
    const float* __restrict__ Ww2, const float* __restrict__ bw2,
    const float* __restrict__ XQ,  const float* __restrict__ XK,
    const float* __restrict__ XV,  float* __restrict__ out)
{
    __shared__ float t3[NSAMP][4];            // ReLU(BN(pr @ Wp1^T))
    __shared__ int   jn[NSAMP];               // neighbor indices
    __shared__ float hbuf[NSAMP][CC + 1];     // relu(bn1(relation)); +1 pad (banks)
    __shared__ float wwb[NSAMP][CSS + 1];     // after Ww1 + bn2 + relu
    __shared__ float softb[NSAMP][CSS + 1];   // softmax weights

    const int tid = threadIdx.x;              // 0..63
    const int i   = blockIdx.x;               // point id

    // --- stage 0: 3-channel positional MLP front-end (one thread per neighbor)
    if (tid < NSAMP) {
        const int j = idx[i * NSAMP + tid];
        jn[tid] = j;
        const float pr0 = p[j * 3 + 0] - p[i * 3 + 0];
        const float pr1 = p[j * 3 + 1] - p[i * 3 + 1];
        const float pr2 = p[j * 3 + 2] - p[i * 3 + 2];
#pragma unroll
        for (int d = 0; d < 3; ++d) {
            float y = pr0 * Wp1[d * 3 + 0] + pr1 * Wp1[d * 3 + 1]
                    + pr2 * Wp1[d * 3 + 2] + bp1[d];
            y = (y - pm[d]) * (pg[d] * rsqrtf(pv[d] + EPSV)) + pb[d];
            t3[tid][d] = fmaxf(y, 0.0f);
        }
    }
    __syncthreads();

    // --- stage 1: positional encoding + relation (thread = channel)
    const int c  = tid;
    const float qc  = XQ[(size_t)i * CC + c];
    const float g1  = wg1[c] * rsqrtf(wv1[c] + EPSV);
    const float m1  = wm1[c], bb1 = wb1[c];
    const float w20 = Wp2[c * 3 + 0], w21 = Wp2[c * 3 + 1],
                w22 = Wp2[c * 3 + 2], pbc = bp2[c];

    float pe[NSAMP], xvr[NSAMP];
#pragma unroll
    for (int ns = 0; ns < NSAMP; ++ns) {
        const int j = jn[ns];
        const float pev = t3[ns][0] * w20 + t3[ns][1] * w21
                        + t3[ns][2] * w22 + pbc;
        pe[ns]  = pev;
        xvr[ns] = XV[(size_t)j * CC + c];                 // L2 gather
        const float w0 = XK[(size_t)j * CC + c] - qc + pev;
        hbuf[ns][c] = fmaxf((w0 - m1) * g1 + bb1, 0.0f);
    }
    __syncthreads();

    // --- stage 2: h[16x64] @ Ww1^T[64x8] as one WMMA pair (wave 0 only)
    if (tid < 32) {
        const int lane = tid;
        const int row  = lane & 15;
        const int kb0  = (lane < 16) ? 0 : 16;
        v16bf a0, a1;
#pragma unroll
        for (int e = 0; e < 16; ++e) {
            const int k = kb0 + e;
            a0[e] = (__bf16)hbuf[row][k];
            a1[e] = (__bf16)hbuf[row][32 + k];
        }
        const int colB = lane & 15;              // cs column (valid < 8)
        const int kbB  = (lane < 16) ? 0 : 16;
        v16bf b0, b1;
#pragma unroll
        for (int e = 0; e < 16; ++e) {
            const float v0 = (colB < CSS) ? Ww1[colB * CC + kbB + e]      : 0.0f;
            const float v1 = (colB < CSS) ? Ww1[colB * CC + 32 + kbB + e] : 0.0f;
            b0[e] = (__bf16)v0;
            b1[e] = (__bf16)v1;
        }
        v8f acc = {};
        acc = __builtin_amdgcn_wmma_f32_16x16x32_bf16(
                  false, a0, false, b0, (short)0, acc, false, false);
        acc = __builtin_amdgcn_wmma_f32_16x16x32_bf16(
                  false, a1, false, b1, (short)0, acc, false, false);
        if (colB < CSS) {
            const float g2  = wg2[colB] * rsqrtf(wv2[colB] + EPSV);
            const float m2  = wm2[colB], bb2 = wb2[colB], bw = bw1[colB];
            const int   mbD = (lane < 16) ? 0 : 8;
#pragma unroll
            for (int r = 0; r < 8; ++r) {
                float v = acc[r] + bw;
                v = fmaxf((v - m2) * g2 + bb2, 0.0f);
                wwb[mbD + r][colB] = v;
            }
        }
    }
    __syncthreads();

    // --- stage 3: 8x8 GEMM (Ww2) + softmax over the 16 neighbors
    if (tid < CSS) {
        float vals[NSAMP];
        float mx = -1e30f;
#pragma unroll
        for (int ns = 0; ns < NSAMP; ++ns) {
            float s = bw2[tid];
#pragma unroll
            for (int cs = 0; cs < CSS; ++cs)
                s += wwb[ns][cs] * Ww2[tid * CSS + cs];
            vals[ns] = s;
            mx = fmaxf(mx, s);
        }
        float den = 0.0f;
#pragma unroll
        for (int ns = 0; ns < NSAMP; ++ns) {
            vals[ns] = __expf(vals[ns] - mx);
            den += vals[ns];
        }
        const float inv = 1.0f / den;
#pragma unroll
        for (int ns = 0; ns < NSAMP; ++ns)
            softb[ns][tid] = vals[ns] * inv;
    }
    __syncthreads();

    // --- stage 4: weighted aggregation (share_planes broadcast of weights)
    const int cs = c & (CSS - 1);
    float o = 0.0f;
#pragma unroll
    for (int ns = 0; ns < NSAMP; ++ns)
        o += (xvr[ns] + pe[ns]) * softb[ns][cs];
    out[(size_t)i * CC + c] = o;
}

// ---------------------------------------------------------------------------
extern "C" void kernel_launch(void* const* d_in, const int* in_sizes, int n_in,
                              void* d_out, int out_size, void* d_ws, size_t ws_size,
                              hipStream_t stream)
{
    const float* p   = (const float*)d_in[0];
    const float* x   = (const float*)d_in[1];
    const int*   idx = (const int*)  d_in[2];
    const float* Wq  = (const float*)d_in[3];  const float* bq  = (const float*)d_in[4];
    const float* Wk  = (const float*)d_in[5];  const float* bk  = (const float*)d_in[6];
    const float* Wv  = (const float*)d_in[7];  const float* bv  = (const float*)d_in[8];
    const float* Wp1 = (const float*)d_in[9];  const float* bp1 = (const float*)d_in[10];
    const float* pg  = (const float*)d_in[11]; const float* pb  = (const float*)d_in[12];
    const float* pm  = (const float*)d_in[13]; const float* pv  = (const float*)d_in[14];
    const float* Wp2 = (const float*)d_in[15]; const float* bp2 = (const float*)d_in[16];
    const float* wg1 = (const float*)d_in[17]; const float* wb1 = (const float*)d_in[18];
    const float* wm1 = (const float*)d_in[19]; const float* wv1 = (const float*)d_in[20];
    const float* Ww1 = (const float*)d_in[21]; const float* bw1 = (const float*)d_in[22];
    const float* wg2 = (const float*)d_in[23]; const float* wb2 = (const float*)d_in[24];
    const float* wm2 = (const float*)d_in[25]; const float* wv2 = (const float*)d_in[26];
    const float* Ww2 = (const float*)d_in[27]; const float* bw2 = (const float*)d_in[28];

    float* XQ = (float*)d_ws;
    float* XK = XQ + (size_t)NPTS * CC;
    float* XV = XK + (size_t)NPTS * CC;

    qkv_wmma<<<NPTS / 16, 32, 0, stream>>>(x, Wq, bq, Wk, bk, Wv, bv, XQ, XK, XV);

    attn_kernel<<<NPTS, 64, 0, stream>>>(
        p, idx, Wp1, bp1, pg, pb, pm, pv, Wp2, bp2,
        wg1, wb1, wm1, wv1, Ww1, bw1, wg2, wb2, wm2, wv2, Ww2, bw2,
        XQ, XK, XV, (float*)d_out);
}
